// YOLOLayer_88536455839775
// MI455X (gfx1250) — matrix operand order repaired and checked by reading, not compiled, not run on hardware
//
#include <hip/hip_runtime.h>

#define NB 32
#define NA 3
#define GHH 76
#define GWW 76
#define NT 48
#define STRIDE_PX 8.0f
#define CELLS_PER_IMG (NA * GHH * GWW)          /* 17328 */
#define TOTAL_CELLS   (NB * CELLS_PER_IMG)      /* 554496 = 2166 * 256 exactly */
#define RECF 16
#define OUT_SCALARS_OFF ((size_t)NB * CELLS_PER_IMG * 6)  /* 3326976 */

typedef __attribute__((ext_vector_type(16))) _Float16 v16h;
typedef __attribute__((ext_vector_type(8)))  float    v8f;

__constant__ float ANC_ALL[9][2] = {
    {10.0f, 13.0f}, {16.0f, 30.0f}, {33.0f, 23.0f},
    {30.0f, 61.0f}, {62.0f, 45.0f}, {59.0f, 119.0f},
    {116.0f, 90.0f}, {156.0f, 198.0f}, {373.0f, 326.0f}};

// ---------------------------------------------------------------------------
// Kernel 0: per-target precompute + per-image target lists + zero scalars.
// rec[t*16]: {timg, best_anchor, gi, gj, valid, tx, ty, tw, th, sc,
//             x0, y0, x1, y1, area, label}
// ---------------------------------------------------------------------------
__global__ void yolo_prep(const float* __restrict__ target,
                          float* __restrict__ rec,
                          int* __restrict__ cnt,
                          int* __restrict__ list,
                          float* __restrict__ out) {
  int t = threadIdx.x;
  if (t < 5) out[OUT_SCALARS_OFF + t] = 0.0f;  // zero loss accumulators
  if (t < NT) {
    const float* tg = target + t * 6;
    float img0 = target[0];
    int timg = (int)(tg[0] - img0);
    float x = tg[1], y = tg[2], w = tg[3], h = tg[4];
    float barea = w * h;
    int bidx = 0;
    float bv = -1.0f;
#pragma unroll
    for (int k = 0; k < 9; ++k) {
      float anw = ANC_ALL[k][0] * (1.0f / 608.0f);
      float anh = ANC_ALL[k][1] * (1.0f / 608.0f);
      float wi = fminf(anw, w) * fminf(anh, h);
      float un = (anw * anh + 1e-16f) + barea - wi;
      float v = wi / un;
      if (v > bv) { bv = v; bidx = k; }   // first max wins (jnp.argmax)
    }
    bool inm = (bidx < NA);               // MASK = {0,1,2}, bi = bidx
    float gx = x * (float)GWW, gy = y * (float)GHH;
    int gi = (int)gx, gj = (int)gy;
    bool valid = inm && gi >= 0 && gj >= 0 && gi < GWW && gj < GHH;
    float ttw = 0.0f, tth = 0.0f;
    if (inm) {
      // log(gw*stride/anchor_px + eps) == log(w*608/anchor_px + eps)
      ttw = __logf(w * 608.0f / ANC_ALL[bidx][0] + 1e-16f);
      tth = __logf(h * 608.0f / ANC_ALL[bidx][1] + 1e-16f);
    }
    float* r = rec + t * RECF;
    r[0]  = (float)timg;
    r[1]  = (float)bidx;
    r[2]  = (float)gi;
    r[3]  = (float)gj;
    r[4]  = valid ? 1.0f : 0.0f;
    r[5]  = gx - floorf(gx);
    r[6]  = gy - floorf(gy);
    r[7]  = ttw;
    r[8]  = tth;
    r[9]  = 2.0f - w * h;
    r[10] = x - w * 0.5f;
    r[11] = y - h * 0.5f;
    r[12] = x + w * 0.5f;
    r[13] = y + h * 0.5f;
    r[14] = barea;
    r[15] = tg[5];
  }
  __syncthreads();
  if (threadIdx.x == 0) {  // preserve target order => last-wins gather semantics
    for (int b = 0; b < NB; ++b) cnt[b] = 0;
    for (int t2 = 0; t2 < NT; ++t2) {
      int b = (int)rec[t2 * RECF + 0];
      if (b >= 0 && b < NB) list[b * NT + cnt[b]++] = t2;
    }
  }
}

// ---------------------------------------------------------------------------
// Kernel 1: one cell per thread; HBM-bound main pass + WMMA wave reduction.
// ---------------------------------------------------------------------------
__global__ __launch_bounds__(256) void yolo_main(const float* __restrict__ x,
                                                 const float* __restrict__ rec,
                                                 const int* __restrict__ cnt,
                                                 const int* __restrict__ list,
                                                 float* __restrict__ out) {
  __shared__ float srec[NT * RECF];
  __shared__ int scnt[NB];
  __shared__ int slist[NB * NT];
  int tid = threadIdx.x;
  for (int q = tid; q < NT * RECF; q += 256) srec[q] = rec[q];
  for (int q = tid; q < NB; q += 256) scnt[q] = cnt[q];
  for (int q = tid; q < NB * NT; q += 256) slist[q] = list[q];
  __syncthreads();

  int id = blockIdx.x * 256 + tid;
  int i = id % GWW;
  int j = (id / GWW) % GHH;
  int a = (id / (GWW * GHH)) % NA;
  int b = id / (GWW * GHH * NA);

  const float* xb = x + (((size_t)b * (NA * 6) + a * 6) * GHH + j) * GWW + i;
  __builtin_prefetch(xb, 0, 3);  // global_prefetch_b8
  const int chs = GHH * GWW;
  float p0 = xb[0 * chs], p1 = xb[1 * chs], p2 = xb[2 * chs];
  float p3 = xb[3 * chs], p4 = xb[4 * chs], p5 = xb[5 * chs];

  float cx = 1.0f / (1.0f + __expf(-p0));
  float cy = 1.0f / (1.0f + __expf(-p1));
  float pconf = 1.0f / (1.0f + __expf(-p4));
  float pcls  = 1.0f / (1.0f + __expf(-p5));
  float saw = ANC_ALL[a][0] * (1.0f / STRIDE_PX);  // scaled anchors
  float sah = ANC_ALL[a][1] * (1.0f / STRIDE_PX);
  float bx = cx + (float)i, by = cy + (float)j;
  float bw = __expf(p2) * saw, bh = __expf(p3) * sah;

  size_t ob = ((size_t)b * CELLS_PER_IMG + (a * GHH + j) * GWW + i) * 6;
  out[ob + 0] = bx * STRIDE_PX;
  out[ob + 1] = by * STRIDE_PX;
  out[ob + 2] = bw * STRIDE_PX;
  out[ob + 3] = bh * STRIDE_PX;
  out[ob + 4] = pconf;
  out[ob + 5] = pcls;

  // normalized predicted box (GH == GW)
  float inv = 1.0f / (float)GWW;
  float bxn = bx * inv, byn = by * inv, bwn = bw * inv, bhn = bh * inv;
  float px0 = bxn - bwn * 0.5f, py0 = byn - bhn * 0.5f;
  float px1 = bxn + bwn * 0.5f, py1 = byn + bhn * 0.5f;
  float parea = bwn * bhn;

  float best = -1.0f;
  int mt = -1;
  int n = scnt[b];
  for (int q = 0; q < n; ++q) {
    int t = slist[b * NT + q];
    const float* r = srec + t * RECF;
    float iw = fmaxf(fminf(r[12], px1) - fmaxf(r[10], px0), 0.0f);
    float ih = fmaxf(fminf(r[13], py1) - fmaxf(r[11], py0), 0.0f);
    float inter = iw * ih;
    best = fmaxf(best, inter / (r[14] + parea - inter + 1e-16f));
    if (r[4] != 0.0f && (int)r[1] == a && (int)r[3] == j && (int)r[2] == i)
      mt = t;  // last matching target wins (scatter .set semantics)
  }

  float s_xy = 0.0f, s_wh = 0.0f, s_conf = 0.0f, s_cls = 0.0f;
  if (mt >= 0) {  // obj cell
    const float* r = srec + mt * RECF;
    float s2 = r[9] * r[9];
    float dx = cx - r[5], dy = cy - r[6];
    s_xy = (dx * dx + dy * dy) * s2;
    float dw = p2 - r[7], dh = p3 - r[8];
    s_wh = (dw * dw + dh * dh) * s2;
    s_conf = -fmaxf(__logf(pconf), -100.0f);  // bce(pconf, 1)
    float tcls = ((int)r[15] == 0) ? 1.0f : 0.0f;
    s_cls = -(tcls * fmaxf(__logf(pcls), -100.0f) +
              (1.0f - tcls) * fmaxf(__logf(1.0f - pcls), -100.0f));
  } else if (best <= 0.5f) {  // noobj, not ignored
    s_conf = -fmaxf(__logf(1.0f - pconf), -100.0f);  // bce(pconf, 0)
  }

  // ---- WMMA 4-component cross-lane reduction (EXEC is full: exact grid) ----
  // A[m,k]: lane m holds K=0..3, lane m+16 holds K=8..11 (16-bit A layout).
  // B[k,n] = 1 for k in {n, n+8}, n<4  =>  D[m,n] = comp_n[m] + comp_n[m+16].
  unsigned lane = (unsigned)(tid & 31);
  v16h av;
#pragma unroll
  for (int e = 0; e < 16; ++e) av[e] = (_Float16)0.0f;
  av[0] = (_Float16)s_xy;
  av[1] = (_Float16)s_wh;
  av[2] = (_Float16)s_conf;
  av[3] = (_Float16)s_cls;
  v16h bsel;
#pragma unroll
  for (int e = 0; e < 16; ++e) {
    bool on = ((lane & 15u) < 4u) && (e == (int)(lane & 15u));
    bsel[e] = on ? (_Float16)1.0f : (_Float16)0.0f;
  }
  v8f cacc;
#pragma unroll
  for (int e = 0; e < 8; ++e) cacc[e] = 0.0f;
  cacc = __builtin_amdgcn_wmma_f32_16x16x32_f16(
      /*neg_a=*/false, av, /*neg_b=*/false, bsel,
      /*c_mod=*/(short)0, cacc, /*reuse_a=*/false, /*reuse_b=*/false);
  float rsum = 0.0f;
#pragma unroll
  for (int e = 0; e < 8; ++e) rsum += cacc[e];   // rows 0..7 (lanes<16) / 8..15
  rsum += __shfl_xor(rsum, 16, 32);              // combine row halves
  if (lane < 4u) {                               // lanes 0..3 hold xy/wh/conf/cls
    atomicAdd(&out[OUT_SCALARS_OFF + 1 + lane], rsum);
    atomicAdd(&out[OUT_SCALARS_OFF + 0], rsum);  // total
  }
}

extern "C" void kernel_launch(void* const* d_in, const int* in_sizes, int n_in,
                              void* d_out, int out_size, void* d_ws, size_t ws_size,
                              hipStream_t stream) {
  const float* x = (const float*)d_in[0];
  const float* target = (const float*)d_in[1];
  float* out = (float*)d_out;
  float* rec = (float*)d_ws;                                  // 768 floats
  int* cnt = (int*)((char*)d_ws + NT * RECF * sizeof(float)); // 32 ints
  int* list = cnt + NB;                                       // 32*48 ints
  yolo_prep<<<1, 64, 0, stream>>>(target, rec, cnt, list, out);
  yolo_main<<<TOTAL_CELLS / 256, 256, 0, stream>>>(x, rec, cnt, list, out);
}
